// GPT_27659589386545
// MI455X (gfx1250) — compile-verified, matrix operand order checked by reading, not run on hardware
//
#include <hip/hip_runtime.h>

// ---------------------------------------------------------------------------
// Types for CDNA5 WMMA (wave32, V_WMMA_F32_16X16X32_BF16)
// ---------------------------------------------------------------------------
typedef __attribute__((ext_vector_type(16))) __bf16 v16bf;
typedef __attribute__((ext_vector_type(8)))  float  v8f;
typedef __attribute__((ext_vector_type(4)))  unsigned int u32x4;
typedef __attribute__((ext_vector_type(8)))  unsigned int u32x8;

union FragBF {
  v16bf v;
  uint4 q[2];     // two 16-byte chunks -> ds_load_b128 x2
};

__device__ __forceinline__ unsigned short f2bf(float f) {
  unsigned int u = __float_as_uint(f);
  u += 0x7FFFu + ((u >> 16) & 1u);      // round-to-nearest-even
  return (unsigned short)(u >> 16);
}

__device__ __forceinline__ v8f wmma_bf16(const FragBF& a, const FragBF& b, v8f c) {
  // (neg_a, A, neg_b, B, c_mod, C, reuse_a, reuse_b)
  return __builtin_amdgcn_wmma_f32_16x16x32_bf16(false, a.v, false, b.v,
                                                 (short)0, c, false, false);
}

__device__ __forceinline__ float gelu_tanh(float x) {
  const float k0 = 0.7978845608028654f;   // sqrt(2/pi)
  const float k1 = 0.044715f;
  float t = tanhf(k0 * (x + k1 * x * x * x));
  return 0.5f * x * (1.0f + t);
}

// ---------------------------------------------------------------------------
// Tensor Data Mover: async 2D bf16 tile load Global -> LDS.
// Builds a D# per cdna5 CDNA5 ISA and issues
// TENSOR_LOAD_TO_LDS (2 SGPR-group form, tensors up to 2D).
// pad_enable inserts 4 DWORDs (8 bf16) after every 16 DWORDs (one 32-elem
// row), reproducing the LDS row stride of 40 bf16 used for bank-conflict
// avoidance -- the DMA engine does the padding, not the VALU.
// ---------------------------------------------------------------------------
__device__ __forceinline__ void tdm_load_2d(unsigned lds_off,
                                            const unsigned short* gsrc,
                                            unsigned k_rem, unsigned n_rem,
                                            unsigned tile_k, unsigned tile_n,
                                            unsigned stride_k)
{
  unsigned long long ga = (unsigned long long)(size_t)gsrc;
  u32x4 g0;
  g0[0] = 1u;                                   // count=1, user descriptor
  g0[1] = lds_off;                              // lds_addr (bytes)
  g0[2] = (unsigned)ga;                         // global_addr[31:0]
  g0[3] = (unsigned)((ga >> 32) & 0x01FFFFFFu)  // global_addr[56:32]
          | (2u << 30);                         // type = 2 ("image")
  u32x8 g1;
  g1[0] = (1u << 16)        // data_size = 2 bytes
        | (1u << 20)        // pad_enable
        | (3u << 22)        // pad_interval: 16 DWORDs (= one 32-elem bf16 row)
        | (3u << 25);       // pad_amount: 4 DWORDs  (-> LDS stride 40 bf16)
  g1[1] = (k_rem & 0xFFFFu) << 16;                       // tensor_dim0[15:0]
  g1[2] = (k_rem >> 16) | ((n_rem & 0xFFFFu) << 16);     // dim0[31:16]|dim1[15:0]
  g1[3] = (n_rem >> 16) | (tile_k << 16);                // dim1[31:16]|tile_dim0
  g1[4] = tile_n & 0xFFFFu;                              // tile_dim1 (tile_dim2=0)
  g1[5] = stride_k;                                      // tensor_dim0_stride[31:0]
  g1[6] = 0u;                                            // stride hi | dim1_stride lo
  g1[7] = 0u;
  asm volatile("tensor_load_to_lds %0, %1" :: "s"(g0), "s"(g1) : "memory");
}

// ---------------------------------------------------------------------------
// GEMM: C[M,N] = act( A[M,K](bf16) * B^T + bias + resid )
//   A: [M,K] bf16 row-major. Bm: [N,K] bf16 row-major (pre-transposed weight).
// Block: 256 threads = 8 waves; tile 128(M) x 64(N); K step 32 (one WMMA).
// Both tiles are fetched by the Tensor Data Mover (wave 0 issues descriptors).
// ---------------------------------------------------------------------------
#define BM 128
#define BN 64
#define BK 32
#define LDT 40   // BK + 8 pad (80-byte rows, 16B aligned, matches TDM pad)

template <bool OUTBF, bool DOGELU>
__global__ __launch_bounds__(256)
void gemm_wmma(const unsigned short* __restrict__ A,
               const unsigned short* __restrict__ Bm,
               const float* __restrict__ bias, const float* __restrict__ resid,
               void* __restrict__ Cout, int M, int N, int K)
{
  __shared__ __align__(16) unsigned short sA[BM][LDT];
  __shared__ __align__(16) unsigned short sB[BN][LDT];   // sB[n][k]

  const int tid  = threadIdx.x;
  const int lane = tid & 31;
  const int wave = tid >> 5;
  const int wm   = wave >> 1;
  const int wn   = wave & 1;
  const int m0   = blockIdx.y * BM;
  const int n0   = blockIdx.x * BN;
  const int half = lane >> 4;
  const int l15  = lane & 15;
  const int kb8  = half * 8;

  const unsigned ldsA = (unsigned)(size_t)&sA[0][0];
  const unsigned ldsB = (unsigned)(size_t)&sB[0][0];

  v8f acc[2][2];
  const v8f vzero = {0.f,0.f,0.f,0.f,0.f,0.f,0.f,0.f};
  for (int i = 0; i < 2; ++i)
    for (int j = 0; j < 2; ++j) acc[i][j] = vzero;

  for (int kk = 0; kk < K; kk += BK) {
    // ---- DMA both tiles into LDS via the Tensor Data Mover ----
    if (wave == 0) {
      tdm_load_2d(ldsA, A  + (size_t)m0 * K + kk,
                  (unsigned)(K - kk), (unsigned)(M - m0), BK, BM, (unsigned)K);
      tdm_load_2d(ldsB, Bm + (size_t)n0 * K + kk,
                  (unsigned)(K - kk), (unsigned)(N - n0), BK, BN, (unsigned)K);
      __builtin_amdgcn_s_wait_tensorcnt(0);
    }
    // L2 warm-up for the next K step while this one computes
    if (kk + BK < K && tid < BM)
      __builtin_prefetch(A + (size_t)(m0 + tid) * K + kk + BK, 0, 1);
    __syncthreads();

    // ---- fragments + WMMA ----
    FragBF af[2], bf[2];
    for (int i = 0; i < 2; ++i) {
      int row = wm * 32 + i * 16 + l15;
      af[i].q[0] = *(const uint4*)&sA[row][kb8];
      af[i].q[1] = *(const uint4*)&sA[row][kb8 + 16];
    }
    for (int j = 0; j < 2; ++j) {
      int col = wn * 32 + j * 16 + l15;
      bf[j].q[0] = *(const uint4*)&sB[col][kb8];
      bf[j].q[1] = *(const uint4*)&sB[col][kb8 + 16];
    }
    for (int i = 0; i < 2; ++i)
      for (int j = 0; j < 2; ++j)
        acc[i][j] = wmma_bf16(af[i], bf[j], acc[i][j]);
    __syncthreads();
  }

  // ---- epilogue: bias / residual / gelu, bounds-checked store ----
  for (int i = 0; i < 2; ++i)
    for (int j = 0; j < 2; ++j)
      for (int r = 0; r < 8; ++r) {
        int row = m0 + wm * 32 + i * 16 + half * 8 + r;
        int col = n0 + wn * 32 + j * 16 + l15;
        if (row < M && col < N) {
          float v = acc[i][j][r];
          if (bias)  v += bias[col];
          if (resid) v += resid[(size_t)row * N + col];
          if (DOGELU) v = gelu_tanh(v);
          if (OUTBF) ((unsigned short*)Cout)[(size_t)row * N + col] = f2bf(v);
          else       ((float*)Cout)[(size_t)row * N + col] = v;
        }
      }
}

// ---------------------------------------------------------------------------
// Flash attention (bf16 qkv in, bf16 y out): per (64 queries, head, batch).
// 128 threads = 4 waves; wave w owns 16 query rows. Online softmax in LDS.
// qkv layout: [B*T, 2304] bf16, col = h*192 + (q:0..63 | k:64..127 | v:128..191)
// ---------------------------------------------------------------------------
__global__ __launch_bounds__(128)
void attn_flash(const unsigned short* __restrict__ qkv,
                unsigned short* __restrict__ y, int T)
{
  __shared__ __align__(16) unsigned short Qs[64][72];
  __shared__ __align__(16) unsigned short Ks[64][72];   // Ks[s][c]
  __shared__ __align__(16) unsigned short Vs[64][72];   // transposed: Vs[d][s]
  __shared__ __align__(16) unsigned short Ps[64][72];   // P bf16
  __shared__ float Ss[64][66];
  __shared__ float mArr[64], lArr[64], fArr[64];

  const int tid  = threadIdx.x;
  const int lane = tid & 31;
  const int w    = tid >> 5;
  const int tb   = blockIdx.x;
  const int h    = blockIdx.y;
  const int b    = blockIdx.z;
  const int half = lane >> 4;
  const int l15  = lane & 15;
  const int kb8  = half * 8;
  const int rowBase = b * T + tb * 64;
  const size_t stride = 2304;
  const float scale = 0.125f;            // 1/sqrt(64)
  const float NEG = -3.0e38f;

  // Load Q tile (64x64 bf16): 512 8-element vectors
  for (int it = tid; it < 512; it += 128) {
    int r  = it >> 3;
    int c8 = (it & 7) * 8;
    *(uint4*)&Qs[r][c8] =
        *(const uint4*)(qkv + (size_t)(rowBase + r) * stride + h * 192 + c8);
  }
  if (tid < 64) { mArr[tid] = NEG; lArr[tid] = 0.f; }
  __syncthreads();

  FragBF qf[2];
  for (int kc = 0; kc < 2; ++kc) {
    int row = w * 16 + l15;
    qf[kc].q[0] = *(const uint4*)&Qs[row][kc * 32 + kb8];
    qf[kc].q[1] = *(const uint4*)&Qs[row][kc * 32 + kb8 + 16];
  }

  const v8f vzero = {0.f,0.f,0.f,0.f,0.f,0.f,0.f,0.f};
  v8f o[4];
  for (int j = 0; j < 4; ++j) o[j] = vzero;

  for (int kbk = 0; kbk <= tb; ++kbk) {
    // Load K (row-major) and V (transposed) bf16 blocks
    for (int it = tid; it < 512; it += 128) {
      int s  = it >> 3;
      int c8 = (it & 7) * 8;
      const unsigned short* kp =
          qkv + (size_t)(b * T + kbk * 64 + s) * stride + h * 192 + 64 + c8;
      *(uint4*)&Ks[s][c8] = *(const uint4*)kp;
      uint4 vv = *(const uint4*)(kp + 64);
      const unsigned short* ve = (const unsigned short*)&vv;
      for (int e = 0; e < 8; ++e) Vs[c8 + e][s] = ve[e];
    }
    __syncthreads();

    // S = Q * K^T  (16 rows x 64 cols per wave)
    v8f sacc[4];
    for (int j = 0; j < 4; ++j) sacc[j] = vzero;
    for (int kc = 0; kc < 2; ++kc)
      for (int j = 0; j < 4; ++j) {
        FragBF kf;
        int col = j * 16 + l15;
        kf.q[0] = *(const uint4*)&Ks[col][kc * 32 + kb8];
        kf.q[1] = *(const uint4*)&Ks[col][kc * 32 + kb8 + 16];
        sacc[j] = wmma_bf16(qf[kc], kf, sacc[j]);
      }
    // scale + causal mask -> LDS
    for (int j = 0; j < 4; ++j)
      for (int r = 0; r < 8; ++r) {
        int qr = w * 16 + half * 8 + r;
        int qg = tb * 64 + qr;
        int kg = kbk * 64 + j * 16 + l15;
        float v = sacc[j][r] * scale;
        if (kg > qg) v = NEG;
        Ss[qr][j * 16 + l15] = v;
      }
    __syncthreads();

    // Online softmax: threads 0..63 each own one query row
    if (tid < 64) {
      int r = tid;
      float mx = Ss[r][0];
      for (int c = 1; c < 64; ++c) mx = fmaxf(mx, Ss[r][c]);
      float mold = mArr[r];
      float mnew = fmaxf(mold, mx);
      float f = __expf(mold - mnew);
      float sum = 0.f;
      for (int c = 0; c < 64; ++c) {
        float p = __expf(Ss[r][c] - mnew);
        sum += p;
        Ps[r][c] = f2bf(p);
      }
      mArr[r] = mnew;
      lArr[r] = lArr[r] * f + sum;
      fArr[r] = f;
    }
    __syncthreads();

    // Rescale O, then O += P * V
    for (int j = 0; j < 4; ++j)
      for (int r = 0; r < 8; ++r)
        o[j][r] *= fArr[w * 16 + half * 8 + r];
    FragBF pf[2];
    for (int kc = 0; kc < 2; ++kc) {
      int row = w * 16 + l15;
      pf[kc].q[0] = *(const uint4*)&Ps[row][kc * 32 + kb8];
      pf[kc].q[1] = *(const uint4*)&Ps[row][kc * 32 + kb8 + 16];
    }
    for (int kc = 0; kc < 2; ++kc)
      for (int j = 0; j < 4; ++j) {
        FragBF vf;
        int col = j * 16 + l15;
        vf.q[0] = *(const uint4*)&Vs[col][kc * 32 + kb8];
        vf.q[1] = *(const uint4*)&Vs[col][kc * 32 + kb8 + 16];
        o[j] = wmma_bf16(pf[kc], vf, o[j]);
      }
    __syncthreads();
  }

  // O / l  -> y[b,t, h*64+d]  (bf16)
  for (int j = 0; j < 4; ++j)
    for (int r = 0; r < 8; ++r) {
      int qr = w * 16 + half * 8 + r;
      float invl = 1.0f / lArr[qr];
      int row = rowBase + qr;
      int col = h * 64 + j * 16 + l15;
      y[(size_t)row * 768 + col] = f2bf(o[j][r] * invl);
    }
}

// ---------------------------------------------------------------------------
// LayerNorm: fp32 in, bf16 out; one block (256 threads) per row
// ---------------------------------------------------------------------------
__global__ __launch_bounds__(256)
void layernorm_k(const float* __restrict__ x, const float* __restrict__ sc,
                 const float* __restrict__ bi, unsigned short* __restrict__ out,
                 int C)
{
  __shared__ float red[256];
  const int tid = threadIdx.x;
  const int row = blockIdx.x;
  const float* xr = x + (size_t)row * C;
  float sum = 0.f, sq = 0.f;
  for (int c = tid; c < C; c += 256) { float v = xr[c]; sum += v; sq += v * v; }

  red[tid] = sum; __syncthreads();
  for (int off = 128; off > 0; off >>= 1) {
    if (tid < off) red[tid] += red[tid + off];
    __syncthreads();
  }
  float mu = red[0] / (float)C;
  __syncthreads();
  red[tid] = sq; __syncthreads();
  for (int off = 128; off > 0; off >>= 1) {
    if (tid < off) red[tid] += red[tid + off];
    __syncthreads();
  }
  float var = red[0] / (float)C - mu * mu;
  float inv = rsqrtf(var + 1e-6f);
  for (int c = tid; c < C; c += 256)
    out[(size_t)row * C + c] = f2bf((xr[c] - mu) * inv * sc[c] + bi[c]);
}

// ---------------------------------------------------------------------------
// Embedding gather (fp32 residual stream): x[row] = wte[idx[row]] + wpe[row%T]
// ---------------------------------------------------------------------------
__global__ __launch_bounds__(256)
void embed_k(const int* __restrict__ idx, const float* __restrict__ wte,
             const float* __restrict__ wpe, float* __restrict__ x, int T, int C)
{
  int row = blockIdx.x;
  int t = row % T;
  int tok = idx[row];
  for (int c = threadIdx.x; c < C; c += 256)
    x[(size_t)row * C + c] = wte[(size_t)tok * C + c] + wpe[(size_t)t * C + c];
}

// ---------------------------------------------------------------------------
// One-time fp32 -> bf16 conversion (straight copy)
// ---------------------------------------------------------------------------
__global__ __launch_bounds__(256)
void conv_bf16_k(const float* __restrict__ in, unsigned short* __restrict__ out,
                 size_t n)
{
  size_t i = (size_t)blockIdx.x * 256 + threadIdx.x;
  size_t stride = (size_t)gridDim.x * 256;
  for (; i < n; i += stride) out[i] = f2bf(in[i]);
}

// ---------------------------------------------------------------------------
// One-time fp32 [K,N] -> bf16 [N,K] transpose-convert (32x32 LDS tiles)
// ---------------------------------------------------------------------------
__global__ __launch_bounds__(256)
void transpose_bf16_k(const float* __restrict__ in,
                      unsigned short* __restrict__ out, int K, int N)
{
  __shared__ float t[32][33];
  int kb = blockIdx.x * 32, nb = blockIdx.y * 32;
  int tx = threadIdx.x & 31, ty = threadIdx.x >> 5;
  for (int i = ty; i < 32; i += 8) {
    int k = kb + i, n = nb + tx;
    t[i][tx] = (k < K && n < N) ? in[(size_t)k * N + n] : 0.f;
  }
  __syncthreads();
  for (int i = ty; i < 32; i += 8) {
    int n = nb + i, k = kb + tx;
    if (n < N && k < K) out[(size_t)n * K + k] = f2bf(t[tx][i]);
  }
}

// ---------------------------------------------------------------------------
// Cross-entropy: one block per token; logsumexp over V, atomicAdd mean NLL
// ---------------------------------------------------------------------------
__global__ void zero_loss(float* p) { if (threadIdx.x == 0 && blockIdx.x == 0) *p = 0.f; }

__global__ __launch_bounds__(256)
void loss_k(const float* __restrict__ logits, const int* __restrict__ targets,
            float* __restrict__ loss, int V, float invNT)
{
  __shared__ float red[256];
  const int tid = threadIdx.x;
  const int row = blockIdx.x;
  const float* lr = logits + (size_t)row * V;
  float mx = -3.0e38f;
  for (int v = tid; v < V; v += 256) mx = fmaxf(mx, lr[v]);
  red[tid] = mx; __syncthreads();
  for (int off = 128; off > 0; off >>= 1) {
    if (tid < off) red[tid] = fmaxf(red[tid], red[tid + off]);
    __syncthreads();
  }
  float m = red[0];
  __syncthreads();
  float s = 0.f;
  for (int v = tid; v < V; v += 256) s += __expf(lr[v] - m);
  red[tid] = s; __syncthreads();
  for (int off = 128; off > 0; off >>= 1) {
    if (tid < off) red[tid] += red[tid + off];
    __syncthreads();
  }
  if (tid == 0) {
    float lse = m + __logf(red[0]);
    float lp = lr[targets[row]] - lse;
    atomicAdd(loss, -lp * invNT);
  }
}

// ---------------------------------------------------------------------------
// Orchestration
// ---------------------------------------------------------------------------
extern "C" void kernel_launch(void* const* d_in, const int* in_sizes, int n_in,
                              void* d_out, int out_size, void* d_ws, size_t ws_size,
                              hipStream_t stream)
{
  (void)in_sizes; (void)n_in; (void)out_size; (void)ws_size;
  const int*   idx     = (const int*)  d_in[0];
  const int*   targets = (const int*)  d_in[1];
  const float* wte     = (const float*)d_in[2];
  const float* wpe     = (const float*)d_in[3];
  const float* ln1_s   = (const float*)d_in[4];
  const float* ln1_b   = (const float*)d_in[5];
  const float* attn_k  = (const float*)d_in[6];
  const float* attn_b  = (const float*)d_in[7];
  const float* aproj_k = (const float*)d_in[8];
  const float* aproj_b = (const float*)d_in[9];
  const float* ln2_s   = (const float*)d_in[10];
  const float* ln2_b   = (const float*)d_in[11];
  const float* fc_k    = (const float*)d_in[12];
  const float* fc_b    = (const float*)d_in[13];
  const float* mproj_k = (const float*)d_in[14];
  const float* mproj_b = (const float*)d_in[15];
  const float* lnf_s   = (const float*)d_in[16];
  const float* lnf_b   = (const float*)d_in[17];

  const int Bb = 2, T = 1024, C = 768, H = 12, Ll = 4, V = 50257;
  const int M = Bb * T;                       // 2048 rows

  // ---- workspace layout ----
  char* wp = (char*)d_ws;
  float* x              = (float*)wp;          wp += (size_t)M * C * 4;   // residual stream
  unsigned short* hbuf  = (unsigned short*)wp; wp += (size_t)M * C * 2;   // LN out (bf16)
  unsigned short* qkvb  = (unsigned short*)wp; wp += (size_t)M * 3 * C * 2;
  unsigned short* ybuf  = (unsigned short*)wp; wp += (size_t)M * C * 2;
  unsigned short* fcb   = (unsigned short*)wp; wp += (size_t)M * 4 * C * 2;
  unsigned short* wqkv  = (unsigned short*)wp; wp += (size_t)Ll * 3 * C * C * 2; // [3C][C]
  unsigned short* wapr  = (unsigned short*)wp; wp += (size_t)Ll * C * C * 2;     // [C][C]
  unsigned short* wfc   = (unsigned short*)wp; wp += (size_t)Ll * 4 * C * C * 2; // [4C][C]
  unsigned short* wmp   = (unsigned short*)wp; wp += (size_t)Ll * C * 4 * C * 2; // [C][4C]
  unsigned short* wteb  = (unsigned short*)wp; wp += (size_t)V * C * 2;          // [V][C]

  float* logits = (float*)d_out;
  float* loss   = logits + (size_t)M * V;

  // ---- one-time weight conversion to bf16 (transposed to [N][K]) ----
  for (int l = 0; l < Ll; ++l) {
    transpose_bf16_k<<<dim3(C / 32, 3 * C / 32), 256, 0, stream>>>(
        attn_k + (size_t)l * C * 3 * C, wqkv + (size_t)l * 3 * C * C, C, 3 * C);
    transpose_bf16_k<<<dim3(C / 32, C / 32), 256, 0, stream>>>(
        aproj_k + (size_t)l * C * C, wapr + (size_t)l * C * C, C, C);
    transpose_bf16_k<<<dim3(C / 32, 4 * C / 32), 256, 0, stream>>>(
        fc_k + (size_t)l * C * 4 * C, wfc + (size_t)l * 4 * C * C, C, 4 * C);
    transpose_bf16_k<<<dim3(4 * C / 32, C / 32), 256, 0, stream>>>(
        mproj_k + (size_t)l * 4 * C * C, wmp + (size_t)l * C * 4 * C, 4 * C, C);
  }
  conv_bf16_k<<<4096, 256, 0, stream>>>(wte, wteb, (size_t)V * C);  // fits in L2 (77MB<192MB)

  embed_k<<<M, 256, 0, stream>>>(idx, wte, wpe, x, T, C);

  for (int l = 0; l < Ll; ++l) {
    layernorm_k<<<M, 256, 0, stream>>>(x, ln1_s + l * C, ln1_b + l * C, hbuf, C);
    gemm_wmma<true, false><<<dim3(3 * C / BN, M / BM), 256, 0, stream>>>(
        hbuf, wqkv + (size_t)l * 3 * C * C, attn_b + l * 3 * C, nullptr,
        qkvb, M, 3 * C, C);
    attn_flash<<<dim3(T / 64, H, Bb), 128, 0, stream>>>(qkvb, ybuf, T);
    gemm_wmma<false, false><<<dim3(C / BN, M / BM), 256, 0, stream>>>(
        ybuf, wapr + (size_t)l * C * C, aproj_b + l * C, x, x, M, C, C);
    layernorm_k<<<M, 256, 0, stream>>>(x, ln2_s + l * C, ln2_b + l * C, hbuf, C);
    gemm_wmma<true, true><<<dim3(4 * C / BN, M / BM), 256, 0, stream>>>(
        hbuf, wfc + (size_t)l * 4 * C * C, fc_b + l * 4 * C, nullptr,
        fcb, M, 4 * C, C);
    gemm_wmma<false, false><<<dim3(C / BN, M / BM), 256, 0, stream>>>(
        fcb, wmp + (size_t)l * C * 4 * C, mproj_b + l * C, x, x, M, C, 4 * C);
  }

  layernorm_k<<<M, 256, 0, stream>>>(x, lnf_s, lnf_b, hbuf, C);
  // logits = hbuf @ wte^T : wteb is already [V][C] = [N][K]
  gemm_wmma<false, false><<<dim3((V + BN - 1) / BN, M / BM), 256, 0, stream>>>(
      hbuf, wteb, nullptr, nullptr, logits, M, V, C);

  zero_loss<<<1, 1, 0, stream>>>(loss);
  loss_k<<<M, 256, 0, stream>>>(logits, targets, loss, V, 1.0f / (float)M);
}
